// QDenseUndirected_6073083756614
// MI455X (gfx1250) — matrix operand order repaired and checked by reading, not compiled, not run on hardware
//
#include <hip/hip_runtime.h>
#include <hip/hip_bf16.h>

typedef _Float16 v16h __attribute__((ext_vector_type(16)));
typedef _Float16 v8h  __attribute__((ext_vector_type(8)));
typedef float    v8f  __attribute__((ext_vector_type(8)));

#define N_WIRES 10
#define DIM     1024          // 2^10 amplitudes
#define PI_F    3.14159265358979323846f
#define LDS_PITCH 40          // halves per staged row (80 B: 16B-aligned, bank-friendly)
#define OUT_PITCH 68          // floats per out-tile row (272 B: 16B-aligned, bank stride 4)

// ---------------------------------------------------------------------------
// Phase A: build the 1024x1024 circuit unitary, one basis column per block.
// State vector lives in LDS (8 KB). Writes U as f16, row-major [m][k].
// ---------------------------------------------------------------------------
__global__ void build_unitary(const float* __restrict__ weights, int qdepth,
                              _Float16* __restrict__ Ure, _Float16* __restrict__ Uim) {
    __shared__ float sre[DIM];
    __shared__ float sim[DIM];
    const int tid = threadIdx.x;      // 256 threads
    const int col = blockIdx.x;       // basis state k

    for (int i = tid; i < DIM; i += 256) {
        sre[i] = (i == col) ? 1.0f : 0.0f;
        sim[i] = 0.0f;
    }
    __syncthreads();

    for (int l = 0; l < qdepth; ++l) {
        // ---- single-qubit Rot on each wire ----
        for (int w = 0; w < N_WIRES; ++w) {
            const float phi   = tanhf(weights[(l * N_WIRES + w) * 3 + 0]) * PI_F;
            const float theta = tanhf(weights[(l * N_WIRES + w) * 3 + 1]) * PI_F;
            const float omega = tanhf(weights[(l * N_WIRES + w) * 3 + 2]) * PI_F;
            const float c = cosf(0.5f * theta);
            const float s = sinf(0.5f * theta);
            const float ap = -0.5f * (phi + omega);   // m00 phase
            const float bp =  0.5f * (phi - omega);   // m01 phase
            const float m00r =  cosf(ap) * c, m00i =  sinf(ap) * c;
            const float m01r = -cosf(bp) * s, m01i = -sinf(bp) * s;
            const float m10r =  cosf(bp) * s, m10i = -sinf(bp) * s;
            const float m11r =  cosf(ap) * c, m11i = -sinf(ap) * c;

            const int p = 9 - w;              // bit position of wire w
            const int h = 1 << p;
            for (int t = tid; t < DIM / 2; t += 256) {
                const int i0 = ((t >> p) << (p + 1)) | (t & (h - 1));
                const int i1 = i0 | h;
                const float s0r = sre[i0], s0i = sim[i0];
                const float s1r = sre[i1], s1i = sim[i1];
                sre[i0] = m00r * s0r - m00i * s0i + m01r * s1r - m01i * s1i;
                sim[i0] = m00r * s0i + m00i * s0r + m01r * s1i + m01i * s1r;
                sre[i1] = m10r * s0r - m10i * s0i + m11r * s1r - m11i * s1i;
                sim[i1] = m10r * s0i + m10i * s0r + m11r * s1i + m11i * s1r;
            }
            __syncthreads();
        }
        // ---- CNOT ring, StronglyEntanglingLayers default range ----
        const int r = (l % (N_WIRES - 1)) + 1;
        for (int w = 0; w < N_WIRES; ++w) {
            const int cm = 1 << (9 - w);
            const int tm = 1 << (9 - ((w + r) % N_WIRES));
            for (int i = tid; i < DIM; i += 256) {
                if ((i & cm) && !(i & tm)) {
                    const int j = i | tm;
                    float tr = sre[i], ti = sim[i];
                    sre[i] = sre[j]; sim[i] = sim[j];
                    sre[j] = tr;     sim[j] = ti;
                }
            }
            __syncthreads();
        }
    }

    for (int m = tid; m < DIM; m += 256) {
        Ure[m * DIM + col] = (_Float16)sre[m];
        Uim[m * DIM + col] = (_Float16)sim[m];
    }
}

// ---------------------------------------------------------------------------
// Phase B: per-row L2 normalization of x, stored as f16 [b][k].
// One wave32 per batch row; 8 rows per 256-thread block.
// Lanes handle f32 pairs -> packed 4B f16 stores (coalesced both ways).
// ---------------------------------------------------------------------------
__global__ void normalize_x(const float* __restrict__ x, _Float16* __restrict__ Xn) {
    const int lane = threadIdx.x & 31;
    const int wave = threadIdx.x >> 5;            // 0..7
    const int b = blockIdx.x * 8 + wave;
    const float* row = x + (size_t)b * DIM;

    float2 v[DIM / 64];
    float ss = 0.0f;
    #pragma unroll
    for (int j = 0; j < DIM / 64; ++j) {
        v[j] = *(const float2*)(row + j * 64 + lane * 2);
        ss += v[j].x * v[j].x + v[j].y * v[j].y;
    }
    #pragma unroll
    for (int m = 16; m >= 1; m >>= 1) ss += __shfl_xor(ss, m, 32);
    const float rn = rsqrtf(ss);

    _Float16* orow = Xn + (size_t)b * DIM;
    #pragma unroll
    for (int j = 0; j < DIM / 64; ++j) {
        union { _Float16 h[2]; unsigned u; } pk;
        pk.h[0] = (_Float16)(v[j].x * rn);
        pk.h[1] = (_Float16)(v[j].y * rn);
        *(unsigned*)(orow + j * 64 + lane * 2) = pk.u;
    }
}

// ---------------------------------------------------------------------------
// Async LDS <-> global 16-byte movers (ASYNCcnt-tracked) + waits.
// LDS operand = low 32 bits of the flat shared pointer (the LDS byte offset).
// ---------------------------------------------------------------------------
__device__ __forceinline__ void async_cp16(const void* gptr, void* lptr) {
    const unsigned lds = (unsigned)(unsigned long long)lptr;
    const unsigned long long ga = (unsigned long long)gptr;
    asm volatile("global_load_async_to_lds_b128 %0, %1, off"
                 :: "v"(lds), "v"(ga) : "memory");
}
__device__ __forceinline__ void async_st16(void* gptr, const void* lptr) {
    const unsigned lds = (unsigned)(unsigned long long)lptr;
    const unsigned long long ga = (unsigned long long)gptr;
    asm volatile("global_store_async_from_lds_b128 %0, %1, off"
                 :: "v"(ga), "v"(lds) : "memory");
}
__device__ __forceinline__ void wait_async_le4() {
    asm volatile("s_wait_asynccnt 0x4" ::: "memory");
}
__device__ __forceinline__ void wait_async_le0() {
    asm volatile("s_wait_asynccnt 0x0" ::: "memory");
}

__device__ __forceinline__ v16h cat8(v8h lo, v8h hi) {
    return __builtin_shufflevector(lo, hi, 0, 1, 2, 3, 4, 5, 6, 7,
                                           8, 9, 10, 11, 12, 13, 14, 15);
}
// A fragment (16x32): lane<16 -> K {0..7,16..23}; lane>=16 -> K {8..15,24..31}
__device__ __forceinline__ v16h frag16x32(const _Float16* rowbase, int lhalf) {
    const v8h lo = *(const v8h*)(rowbase + lhalf * 8);
    const v8h hi = *(const v8h*)(rowbase + 16 + lhalf * 8);
    return cat8(lo, hi);
}

// ---------------------------------------------------------------------------
// Phase C: probs = clip(1024 * (|Re(U)X|^2 + |Im(U)X|^2), 0, 1) via WMMA.
// Block = 8 waves as 2(M) x 4(N); wave tile 32x32 (2x2 frags x {re,im}).
// K slices of 32 staged through LDS by async copies (double-buffered);
// output tile staged in LDS and drained with async LDS->global stores.
// ---------------------------------------------------------------------------
__global__ void gemm_prob(const _Float16* __restrict__ Ure,
                          const _Float16* __restrict__ Uim,
                          const _Float16* __restrict__ Xn,
                          float* __restrict__ out) {
    // 40960 B shared pool: staging buffers during the K loop,
    // reused as the 128x68 f32 output tile (34816 B) in the epilogue.
    __shared__ __align__(16) unsigned char smem[40960];
    _Float16* sAre = (_Float16*)smem;                        // [2][64*LDS_PITCH]
    _Float16* sAim = sAre + 2 * 64 * LDS_PITCH;              // [2][64*LDS_PITCH]
    _Float16* sB   = sAim + 2 * 64 * LDS_PITCH;              // [2][128*LDS_PITCH]
    float*    sOut = (float*)smem;                           // [128][OUT_PITCH]

    const int tid   = threadIdx.x;
    const int lane  = tid & 31;
    const int wave  = tid >> 5;           // 0..7
    const int wm    = wave & 1;           // 2 waves along M
    const int wn    = wave >> 1;          // 4 waves along N
    const int lhalf = lane >> 4;
    const int lmod  = lane & 15;

    const int mblk = blockIdx.x * 64;
    const int nblk = blockIdx.y * 128;

    // Staging map: thread -> (row, 16B chunk)
    const int crow = tid >> 2;            // 0..63
    const int ccol = (tid & 3) * 8;       // halves 0/8/16/24

    v8f acc_re[2][2] = {};
    v8f acc_im[2][2] = {};

    // Issue the async copies for K slice `k` into buffer `buf` (4 ops/thread).
    auto issue = [&](int buf, int k) {
        async_cp16(Ure + (size_t)(mblk + crow) * DIM + k + ccol,
                   &sAre[buf * 64 * LDS_PITCH + crow * LDS_PITCH + ccol]);
        async_cp16(Uim + (size_t)(mblk + crow) * DIM + k + ccol,
                   &sAim[buf * 64 * LDS_PITCH + crow * LDS_PITCH + ccol]);
        #pragma unroll
        for (int j = 0; j < 2; ++j) {     // 128 X rows = 512 chunks / 256 thr
            const int c  = tid + 256 * j;
            const int rb = c >> 2;
            const int cb = (c & 3) * 8;
            async_cp16(Xn + (size_t)(nblk + rb) * DIM + k + cb,
                       &sB[buf * 128 * LDS_PITCH + rb * LDS_PITCH + cb]);
        }
    };

    issue(0, 0);

    const int NSTEP = DIM / 32;           // 32
    for (int i = 0; i < NSTEP; ++i) {
        const int cur = i & 1;
        if (i + 1 < NSTEP) {
            issue(cur ^ 1, (i + 1) * 32); // prefetch next slice
            wait_async_le4();             // previous slice (oldest 4) complete
        } else {
            wait_async_le0();
        }
        __syncthreads();                  // everyone's copies visible

        // Fragments from LDS
        v16h are[2], aim[2], bf[2];
        #pragma unroll
        for (int mf = 0; mf < 2; ++mf) {
            const int row = wm * 32 + mf * 16 + lmod;
            are[mf] = frag16x32(&sAre[cur * 64 * LDS_PITCH + row * LDS_PITCH], lhalf);
            aim[mf] = frag16x32(&sAim[cur * 64 * LDS_PITCH + row * LDS_PITCH], lhalf);
        }
        #pragma unroll
        for (int nf = 0; nf < 2; ++nf) {
            const int col = wn * 32 + nf * 16 + lmod;
            const _Float16* bp = &sB[cur * 128 * LDS_PITCH + col * LDS_PITCH + lhalf * 8];
            bf[nf] = cat8(*(const v8h*)(bp), *(const v8h*)(bp + 16));
        }
        #pragma unroll
        for (int mf = 0; mf < 2; ++mf)
            #pragma unroll
            for (int nf = 0; nf < 2; ++nf) {
                acc_re[mf][nf] = __builtin_amdgcn_wmma_f32_16x16x32_f16(
                    false, are[mf], false, bf[nf], (short)0, acc_re[mf][nf],
                    false, false);
                acc_im[mf][nf] = __builtin_amdgcn_wmma_f32_16x16x32_f16(
                    false, aim[mf], false, bf[nf], (short)0, acc_im[mf][nf],
                    false, false);
            }
        __syncthreads();                  // readers done before next overwrite
    }

    // Epilogue 1: probs into the LDS out-tile [bLocal][mLocal].
    // C/D layout — vgpr j: lanes 0-15 -> M=j, lanes 16-31 -> M=j+8.
    #pragma unroll
    for (int mf = 0; mf < 2; ++mf)
        #pragma unroll
        for (int nf = 0; nf < 2; ++nf) {
            const int bl = wn * 32 + nf * 16 + lmod;
            #pragma unroll
            for (int j = 0; j < 8; ++j) {
                const int ml = wm * 32 + mf * 16 + j + lhalf * 8;
                const float re = acc_re[mf][nf][j];
                const float im = acc_im[mf][nf][j];
                float p = (re * re + im * im) * 1024.0f;
                p = fminf(fmaxf(p, 0.0f), 1.0f);
                sOut[bl * OUT_PITCH + ml] = p;
            }
        }
    __syncthreads();

    // Epilogue 2: drain tile with coalesced async LDS->global stores.
    // 128 rows x 64 floats = 2048 x 16B chunks; 8 per thread.
    #pragma unroll
    for (int p2 = 0; p2 < 8; ++p2) {
        const int c   = tid + 256 * p2;
        const int row = c >> 4;           // 0..127 (batch-local)
        const int q   = (c & 15) * 4;     // float offset within row
        async_st16(out + (size_t)(nblk + row) * DIM + mblk + q,
                   &sOut[row * OUT_PITCH + q]);
    }
    wait_async_le0();                     // stores complete before wave exit
}

// ---------------------------------------------------------------------------
extern "C" void kernel_launch(void* const* d_in, const int* in_sizes, int n_in,
                              void* d_out, int out_size, void* d_ws, size_t ws_size,
                              hipStream_t stream) {
    const float* x = (const float*)d_in[0];       // (B,1,32,32) f32
    const float* w = (const float*)d_in[1];       // (qdepth,10,3) f32
    const int B      = in_sizes[0] / DIM;         // 8192
    const int qdepth = in_sizes[1] / (N_WIRES * 3);

    unsigned char* ws = (unsigned char*)d_ws;
    _Float16* Ure = (_Float16*)(ws);                                   // 2 MB
    _Float16* Uim = (_Float16*)(ws + (size_t)DIM * DIM * 2);           // 2 MB
    _Float16* Xn  = (_Float16*)(ws + (size_t)DIM * DIM * 4);           // 16 MB

    build_unitary<<<DIM, 256, 0, stream>>>(w, qdepth, Ure, Uim);
    normalize_x<<<B / 8, 256, 0, stream>>>(x, Xn);
    gemm_prob<<<dim3(DIM / 64, B / 128), 256, 0, stream>>>(Ure, Uim, Xn,
                                                           (float*)d_out);
}
